// SparseAttention_40089224741104
// MI455X (gfx1250) — compile-verified, matrix operand order checked by reading, not run on hardware
//
#include <hip/hip_runtime.h>

// ---------------------------------------------------------------------------
// Sparse attention for MI455X (gfx1250, wave32, WMMA + Tensor Data Mover).
// All matmuls run on v_wmma_f32_16x16x32_bf16 (f32 accumulate).
// GEMM tiles are DMA'd into LDS with TENSOR_LOAD_TO_LDS, double-buffered
// against TENSORcnt.
// ---------------------------------------------------------------------------

#define S_LEN 2048
#define HID   1024
#define NH    16
#define HD    64
#define HALF_WIN 512     // LOCAL_WINDOW/2
#define GSTRIDE_MASK 511 // GLOBAL_STRIDE-1 (power of two)

typedef __attribute__((ext_vector_type(16))) __bf16 v16bf;
typedef __attribute__((ext_vector_type(8)))  float  v8f;
typedef __attribute__((ext_vector_type(4)))  unsigned int v4u;
typedef __attribute__((ext_vector_type(8)))  int v8i;
typedef __attribute__((ext_vector_type(4)))  int v4i;

union ABFrag { uint4 u[2]; v16bf v; };

static __device__ __forceinline__ v8f wmma_bf16(v16bf a, v16bf b, v8f c) {
  // (neg_a, A, neg_b, B, c_mod, C, reuse_a, reuse_b)
  return __builtin_amdgcn_wmma_f32_16x16x32_bf16(false, a, false, b, (short)0, c,
                                                 false, false);
}

static __device__ __forceinline__ v8f zero8() {
  v8f z;
#pragma unroll
  for (int i = 0; i < 8; ++i) z[i] = 0.0f;
  return z;
}

// ---------------------------------------------------------------------------
// Tensor Data Mover: 2D tile load Global -> LDS (data_size = 4 bytes).
// D# layout per CDNA5 ISA ch.8: group0 = {flags/count, lds_addr, gaddr_lo,
// gaddr_hi|type}, group1 = dims/strides. Groups 2/3 zero (2D tensor).
// ---------------------------------------------------------------------------
static __device__ __forceinline__ void tdm_load_2d(
    unsigned lds_off, const void* gptr,
    unsigned tile_d0_dw, unsigned tile_d1,
    unsigned tensor_d0_dw, unsigned tensor_d1, unsigned stride_dw) {
  unsigned long long ga = (unsigned long long)gptr;
  v4u g0;
  g0[0] = 1u;                                   // count=1, user mode, no gather
  g0[1] = lds_off;                              // LDS byte address
  g0[2] = (unsigned)ga;                         // global addr [31:0]
  g0[3] = (unsigned)((ga >> 32) & 0x01FFFFFFu)  // global addr [56:32]
          | (2u << 30);                         // type = 2 ("image")
  v8i g1;
  g1[0] = (int)(2u << 16);                      // wg_mask=0, data_size=2 (4B)
  g1[1] = (int)((tensor_d0_dw & 0xFFFFu) << 16);          // tensor_dim0 lo16
  g1[2] = (int)((tensor_d0_dw >> 16) | ((tensor_d1 & 0xFFFFu) << 16));
  g1[3] = (int)((tensor_d1 >> 16) | (tile_d0_dw << 16));  // tile_dim0
  g1[4] = (int)(tile_d1 & 0xFFFFu);             // tile_dim1 (tile_dim2 = 0)
  g1[5] = (int)stride_dw;                       // tensor_dim0_stride [31:0]
  g1[6] = 0;                                    // stride hi, dim1_stride lo
  g1[7] = 0;
  v4i z4 = {0, 0, 0, 0};
#if __clang_major__ >= 23
  v8i z8 = {0, 0, 0, 0, 0, 0, 0, 0};
  __builtin_amdgcn_tensor_load_to_lds(g0, g1, z4, z4, z8, 0);
#else
  __builtin_amdgcn_tensor_load_to_lds(g0, g1, z4, z4, 0);
#endif
}

static __device__ __forceinline__ unsigned lds_byte_addr(const void* p) {
  // LDS flat addresses keep the LDS byte offset in the low 32 bits.
  return (unsigned)(unsigned long long)p;
}

// ---------------------------------------------------------------------------
// Prep kernels (memory-bound, tiny fraction of runtime)
// ---------------------------------------------------------------------------
__global__ __launch_bounds__(256) void convert_f32_to_bf16(
    const float* __restrict__ in, __bf16* __restrict__ out, int n) {
  int i = blockIdx.x * 256 + threadIdx.x;
  if (i < n) out[i] = (__bf16)in[i];
}

// out[c*R + r] = (bf16) in[r*C + c]
__global__ __launch_bounds__(256) void transpose_f32_to_bf16(
    const float* __restrict__ in, __bf16* __restrict__ out, int R, int C) {
  __shared__ float tile[32][33];
  int bx = blockIdx.x * 32, by = blockIdx.y * 32;
  int tx = threadIdx.x & 31, ty = threadIdx.x >> 5;
#pragma unroll
  for (int yy = ty; yy < 32; yy += 8)
    tile[yy][tx] = in[(size_t)(by + yy) * C + bx + tx];
  __syncthreads();
#pragma unroll
  for (int yy = ty; yy < 32; yy += 8)
    out[(size_t)(bx + yy) * R + by + tx] = (__bf16)tile[tx][yy];
}

// out[c*R + r] = in[r*C + c]   (bf16 -> bf16)
__global__ __launch_bounds__(256) void transpose_bf16(
    const __bf16* __restrict__ in, __bf16* __restrict__ out, int R, int C) {
  __shared__ __bf16 tile[32][33];
  int bx = blockIdx.x * 32, by = blockIdx.y * 32;
  int tx = threadIdx.x & 31, ty = threadIdx.x >> 5;
#pragma unroll
  for (int yy = ty; yy < 32; yy += 8)
    tile[yy][tx] = in[(size_t)(by + yy) * C + bx + tx];
  __syncthreads();
#pragma unroll
  for (int yy = ty; yy < 32; yy += 8)
    out[(size_t)(bx + yy) * R + by + tx] = tile[tx][yy];
}

// ---------------------------------------------------------------------------
// bf16 GEMM: C[M,N] = A[M,K] * B[K,N], with B supplied transposed (BT[N,K]).
// 128x128 block tile, BK=32, 8 waves; wave computes 64x32 via 4x2 WMMA accums.
// Tiles arrive in LDS via TDM (TENSOR_LOAD_TO_LDS), double-buffered.
// ---------------------------------------------------------------------------
#define BM 128
#define BN 128
#define BK 32

template <bool F32OUT>
__global__ __launch_bounds__(256) void gemm_bf16(
    const __bf16* __restrict__ A, const __bf16* __restrict__ BT,
    void* __restrict__ Cout, int M, int N, int K) {
  __shared__ __align__(16) __bf16 As[2][BM * BK];
  __shared__ __align__(16) __bf16 Bs[2][BN * BK];

  const int tid  = threadIdx.x;
  const int lane = tid & 31;
  const int g    = lane >> 4;   // lane group (K-run selector)
  const int ml   = lane & 15;   // row/col within 16
  const int wid  = __builtin_amdgcn_readfirstlane(tid >> 5);  // scalar wave id
  const int waveM = wid & 1;    // 0..1 -> 64-row slab
  const int waveN = wid >> 1;   // 0..3 -> 32-col slab
  const int m0 = blockIdx.y * BM;
  const int n0 = blockIdx.x * BN;

  v8f acc[4][2];
#pragma unroll
  for (int i = 0; i < 4; ++i)
#pragma unroll
    for (int j = 0; j < 2; ++j) acc[i][j] = zero8();

  const int kTiles = K / BK;
  const unsigned kdw = (unsigned)(K / 2);  // row length in dwords

  // prologue: DMA tiles for kt=0 into buffer 0
  if (wid == 0) {
    tdm_load_2d(lds_byte_addr(&As[0][0]), A + (size_t)m0 * K,
                BK / 2, BM, kdw, (unsigned)M, kdw);
    tdm_load_2d(lds_byte_addr(&Bs[0][0]), BT + (size_t)n0 * K,
                BK / 2, BN, kdw, (unsigned)N, kdw);
  }

  int buf = 0;
  for (int kt = 0; kt < kTiles; ++kt, buf ^= 1) {
    if (wid == 0) {
      if (kt + 1 < kTiles) {
        // issue next tiles, then wait until the two older DMAs finished
        tdm_load_2d(lds_byte_addr(&As[buf ^ 1][0]),
                    A + (size_t)m0 * K + (size_t)(kt + 1) * BK,
                    BK / 2, BM, kdw, (unsigned)M, kdw);
        tdm_load_2d(lds_byte_addr(&Bs[buf ^ 1][0]),
                    BT + (size_t)n0 * K + (size_t)(kt + 1) * BK,
                    BK / 2, BN, kdw, (unsigned)N, kdw);
        __builtin_amdgcn_s_wait_tensorcnt(2);
      } else {
        __builtin_amdgcn_s_wait_tensorcnt(0);
      }
    }
    __syncthreads();

    const __bf16* as = &As[buf][0];
    const __bf16* bs = &Bs[buf][0];
    ABFrag af[4], bfr[2];
#pragma unroll
    for (int i = 0; i < 4; ++i) {
      int r = waveM * 64 + i * 16 + ml;
      af[i].u[0] = *(const uint4*)(&as[r * BK + g * 8]);
      af[i].u[1] = *(const uint4*)(&as[r * BK + 16 + g * 8]);
    }
#pragma unroll
    for (int j = 0; j < 2; ++j) {
      int r = waveN * 32 + j * 16 + ml;
      bfr[j].u[0] = *(const uint4*)(&bs[r * BK + g * 8]);
      bfr[j].u[1] = *(const uint4*)(&bs[r * BK + 16 + g * 8]);
    }
#pragma unroll
    for (int i = 0; i < 4; ++i)
#pragma unroll
      for (int j = 0; j < 2; ++j)
        acc[i][j] = wmma_bf16(af[i].v, bfr[j].v, acc[i][j]);
    __syncthreads();
  }

  // C/D layout: lane n = ml (col), VGPR r -> row r + 8*g
#pragma unroll
  for (int i = 0; i < 4; ++i)
#pragma unroll
    for (int j = 0; j < 2; ++j)
#pragma unroll
      for (int r = 0; r < 8; ++r) {
        int row = m0 + waveM * 64 + i * 16 + r + 8 * g;
        int col = n0 + waveN * 32 + j * 16 + ml;
        float v = acc[i][j][r];
        if (F32OUT)
          ((float*)Cout)[(size_t)row * N + col] = v;
        else
          ((__bf16*)Cout)[(size_t)row * N + col] = (__bf16)v;
      }
}

// ---------------------------------------------------------------------------
// Sparse flash attention. One wave handles 16 query rows of one head.
// Qb,Kb: bf16 [S][H*D] row-major. VT: bf16 [H*D][S]. Ob: bf16 [S][H*D].
// ---------------------------------------------------------------------------
__global__ __launch_bounds__(128) void attn_sparse(
    const __bf16* __restrict__ Qb, const __bf16* __restrict__ Kb,
    const __bf16* __restrict__ VT, __bf16* __restrict__ Ob) {
  __shared__ __align__(16) __bf16 Pst[4][16 * 32];  // per-wave P staging

  const int lane = threadIdx.x & 31;
  const int wv   = threadIdx.x >> 5;
  const int g    = lane >> 4;
  const int ml   = lane & 15;
  const int h    = blockIdx.y;
  // scalarize: wave-uniform by construction; readfirstlane makes the sparse
  // tile-skip branch a pure SALU branch so EXEC stays all-ones around WMMA.
  const int q0 = __builtin_amdgcn_readfirstlane((blockIdx.x * 4 + wv) * 16);

  // Q A-fragments (d 0..31 and 32..63), reused over all key tiles
  const __bf16* qrow = Qb + (size_t)(q0 + ml) * HID + h * HD;
  ABFrag aq[2];
  aq[0].u[0] = *(const uint4*)(qrow + 8 * g);
  aq[0].u[1] = *(const uint4*)(qrow + 16 + 8 * g);
  aq[1].u[0] = *(const uint4*)(qrow + 32 + 8 * g);
  aq[1].u[1] = *(const uint4*)(qrow + 48 + 8 * g);

  v8f oacc[4];
#pragma unroll
  for (int t = 0; t < 4; ++t) oacc[t] = zero8();
  float mrow[8], lrow[8];
#pragma unroll
  for (int r = 0; r < 8; ++r) { mrow[r] = -3.0e38f; lrow[r] = 0.0f; }

  const float sc = 0.125f;  // 1/sqrt(64)
  const int jEnd = q0 + 16; // causal bound for this q tile

  for (int j0 = 0; j0 < jEnd; j0 += 32) {
    // structural sparsity (scalar): keep tile only if it touches the local
    // band or contains a stride-512 global column
    bool anyLocal = (j0 + 31) >= (q0 - HALF_WIN);
    bool anyGlob  = (j0 & GSTRIDE_MASK) == 0;
    if (!(anyLocal || anyGlob)) continue;

    // ---- scores: 2 column subtiles x (2 chained WMMAs over d) ----
    v8f s[2];
#pragma unroll
    for (int t = 0; t < 2; ++t) {
      const __bf16* krow = Kb + (size_t)(j0 + t * 16 + ml) * HID + h * HD;
      ABFrag b0, b1;
      b0.u[0] = *(const uint4*)(krow + 8 * g);
      b0.u[1] = *(const uint4*)(krow + 16 + 8 * g);
      b1.u[0] = *(const uint4*)(krow + 32 + 8 * g);
      b1.u[1] = *(const uint4*)(krow + 48 + 8 * g);
      v8f z = zero8();
      z = wmma_bf16(aq[0].v, b0.v, z);
      z = wmma_bf16(aq[1].v, b1.v, z);
      s[t] = z;
    }

    // ---- scale + mask ----
#pragma unroll
    for (int t = 0; t < 2; ++t)
#pragma unroll
      for (int r = 0; r < 8; ++r) {
        int i = q0 + r + 8 * g;
        int j = j0 + t * 16 + ml;
        float v = s[t][r] * sc;
        bool ok = (j <= i) && (((i - j) <= HALF_WIN) || ((j & GSTRIDE_MASK) == 0));
        s[t][r] = ok ? v : -1.0e30f;
      }

    // ---- online softmax (row = r + 8g, reduce across 16-lane half) ----
#pragma unroll
    for (int r = 0; r < 8; ++r) {
      float v = fmaxf(s[0][r], s[1][r]);
#pragma unroll
      for (int m = 1; m < 16; m <<= 1) v = fmaxf(v, __shfl_xor(v, m, 32));
      float mn   = fmaxf(mrow[r], v);
      float corr = __expf(mrow[r] - mn);
      float p0   = __expf(s[0][r] - mn);
      float p1   = __expf(s[1][r] - mn);
      float rs   = p0 + p1;
#pragma unroll
      for (int m = 1; m < 16; m <<= 1) rs += __shfl_xor(rs, m, 32);
      lrow[r] = lrow[r] * corr + rs;
      mrow[r] = mn;
      s[0][r] = p0;
      s[1][r] = p1;
#pragma unroll
      for (int t = 0; t < 4; ++t) oacc[t][r] *= corr;
    }

    // ---- re-layout P (C layout -> A fragment) through LDS ----
    __bf16* lp = Pst[wv];
#pragma unroll
    for (int r = 0; r < 8; ++r) {
      lp[(r + 8 * g) * 32 + ml]      = (__bf16)s[0][r];
      lp[(r + 8 * g) * 32 + 16 + ml] = (__bf16)s[1][r];
    }
    ABFrag pa;
    pa.u[0] = *(const uint4*)(&lp[ml * 32 + 8 * g]);
    pa.u[1] = *(const uint4*)(&lp[ml * 32 + 16 + 8 * g]);

    // ---- P @ V : 4 d-subtiles, K = 32 keys ----
#pragma unroll
    for (int t = 0; t < 4; ++t) {
      const __bf16* vrow = VT + (size_t)(h * HD + t * 16 + ml) * S_LEN + j0;
      ABFrag bv;
      bv.u[0] = *(const uint4*)(vrow + 8 * g);
      bv.u[1] = *(const uint4*)(vrow + 16 + 8 * g);
      oacc[t] = wmma_bf16(pa.v, bv.v, oacc[t]);
    }
  }

  // ---- normalize and write ----
#pragma unroll
  for (int t = 0; t < 4; ++t)
#pragma unroll
    for (int r = 0; r < 8; ++r) {
      float v = oacc[t][r] / lrow[r];
      Ob[(size_t)(q0 + r + 8 * g) * HID + h * HD + t * 16 + ml] = (__bf16)v;
    }
}

// ---------------------------------------------------------------------------
// Host-side orchestration
// ---------------------------------------------------------------------------
extern "C" void kernel_launch(void* const* d_in, const int* in_sizes, int n_in,
                              void* d_out, int out_size, void* d_ws, size_t ws_size,
                              hipStream_t stream) {
  const float* X  = (const float*)d_in[0];
  const float* Wq = (const float*)d_in[1];
  const float* Wk = (const float*)d_in[2];
  const float* Wv = (const float*)d_in[3];
  const float* Wo = (const float*)d_in[4];
  float* out = (float*)d_out;

  char* ws = (char*)d_ws;
  const size_t SZ_X  = (size_t)S_LEN * HID * 2;  // 4 MB bf16
  const size_t SZ_W  = (size_t)HID * HID * 2;    // 2 MB bf16
  __bf16* Xb  = (__bf16*)(ws);
  __bf16* WqT = (__bf16*)(ws + SZ_X);
  __bf16* WkT = (__bf16*)(ws + SZ_X + SZ_W);
  __bf16* WvT = (__bf16*)(ws + SZ_X + 2 * SZ_W);
  __bf16* WoT = (__bf16*)(ws + SZ_X + 3 * SZ_W);
  __bf16* Qb  = (__bf16*)(ws + SZ_X + 4 * SZ_W);
  __bf16* Kb  = (__bf16*)(ws + 2 * SZ_X + 4 * SZ_W);
  __bf16* Vb  = (__bf16*)(ws + 3 * SZ_X + 4 * SZ_W);
  __bf16* Vt  = (__bf16*)(ws + 4 * SZ_X + 4 * SZ_W);
  __bf16* Ob  = (__bf16*)(ws + 5 * SZ_X + 4 * SZ_W);
  // total ws usage: 6*4MB + 4*2MB = 32 MB

  const int nX = S_LEN * HID;
  convert_f32_to_bf16<<<(nX + 255) / 256, 256, 0, stream>>>(X, Xb, nX);

  dim3 tgridW(HID / 32, HID / 32);
  transpose_f32_to_bf16<<<tgridW, 256, 0, stream>>>(Wq, WqT, HID, HID);
  transpose_f32_to_bf16<<<tgridW, 256, 0, stream>>>(Wk, WkT, HID, HID);
  transpose_f32_to_bf16<<<tgridW, 256, 0, stream>>>(Wv, WvT, HID, HID);
  transpose_f32_to_bf16<<<tgridW, 256, 0, stream>>>(Wo, WoT, HID, HID);

  dim3 ggrid(HID / BN, S_LEN / BM);
  gemm_bf16<false><<<ggrid, 256, 0, stream>>>(Xb, WqT, (void*)Qb, S_LEN, HID, HID);
  gemm_bf16<false><<<ggrid, 256, 0, stream>>>(Xb, WkT, (void*)Kb, S_LEN, HID, HID);
  gemm_bf16<false><<<ggrid, 256, 0, stream>>>(Xb, WvT, (void*)Vb, S_LEN, HID, HID);

  dim3 tgridV(HID / 32, S_LEN / 32);
  transpose_bf16<<<tgridV, 256, 0, stream>>>(Vb, Vt, S_LEN, HID);

  dim3 agrid(S_LEN / 16 / 4, NH);  // 32 x 16 blocks, 4 waves/block
  attn_sparse<<<agrid, 128, 0, stream>>>(Qb, Kb, Vt, Ob);

  gemm_bf16<true><<<ggrid, 256, 0, stream>>>(Ob, WoT, (void*)out, S_LEN, HID, HID);
}
